// MoELayer_63496796504387
// MI455X (gfx1250) — matrix operand order repaired and checked by reading, not compiled, not run on hardware
//
#include <hip/hip_runtime.h>
#include <cstdint>

// ---------------------------------------------------------------------------
// MoE SwiGLU top-2 for MI455X (gfx1250, wave32, WMMA bf16 16x16x32).
//  - router + atomic compaction (order-independent => deterministic output)
//  - dom[e] folded into fp32 biases b1/b3 (exact: silu((x+d)W1)=silu(xW1+dW1))
//  - weights pre-packed f32->bf16 into WMMA B-fragment layout (1KB/frag)
//  - fused GEMM1(SwiGLU)->GEMM2 per 128-token CTA, out acc in VGPRs,
//    combine via global_atomic_add_f32 (exactly 2 adds onto 0.0 => determ.)
//  - per-m-tile LDS base pointers precomputed so all ds_loads use short
//    immediate offsets (DS imm is 16-bit; LDS block is 300KB) -> no per-step
//    VALU address math between WMMAs, no WMMA->VALU hazard NOPs.
// ---------------------------------------------------------------------------

#define HDIM 1024
#define IDIM 4096
#define NEXP 8
#define TMAX 8192
#define MTILE 128
#define CCHUNK 128
#define NCHUNKS (IDIM / CCHUNK)   // 32
#define KT_H (HDIM / 32)          // 32 k-tiles for GEMM1
#define KT_I (IDIM / 32)          // 128 k-tiles for GEMM2
#define XSTR (HDIM + 8)           // LDS row stride (bf16 elems), bank-skewed
#define HSTR (CCHUNK + 8)
#define PACKQ ((size_t)NEXP * HDIM * IDIM * 2 / 16)  // uint4 per packed matrix

typedef unsigned short u16;
typedef __attribute__((ext_vector_type(16))) __bf16 v16bf;
typedef __attribute__((ext_vector_type(8)))  float  v8f;

struct Pack32 { uint4 lo, hi; };

static __device__ __forceinline__ u16 f2bf(float f) {   // RNE f32->bf16
  unsigned int u = __builtin_bit_cast(unsigned int, f);
  u += 0x7FFFu + ((u >> 16) & 1u);
  return (u16)(u >> 16);
}
static __device__ __forceinline__ v16bf load_frag_g(const uint4* p) {
  Pack32 t; t.lo = p[0]; t.hi = p[1];
  return __builtin_bit_cast(v16bf, t);
}
// A-fragment from precomputed per-tile LDS pointer + small elem offset
static __device__ __forceinline__ v16bf load_frag_a(const u16* p, int eoff) {
  Pack32 t;
  t.lo = *(const uint4*)(p + eoff);
  t.hi = *(const uint4*)(p + eoff + 16);
  return __builtin_bit_cast(v16bf, t);
}
static __device__ __forceinline__ v8f wmma_b(v16bf a, v16bf b, v8f c) {
  return __builtin_amdgcn_wmma_f32_16x16x32_bf16(false, a, false, b, (short)0, c,
                                                 false, false);
}

// ---------------------------------------------------------------------------
__global__ void moe_zero_kernel(float* __restrict__ out, int n, int* __restrict__ cnt) {
  int i = blockIdx.x * blockDim.x + threadIdx.x;
  if (i < NEXP) cnt[i] = 0;
  for (int j = i; j < n; j += gridDim.x * blockDim.x) out[j] = 0.f;
}

// router: one wave per token, top-2 + softmax, atomic compaction per expert
__global__ void moe_router_kernel(const float* __restrict__ x,
                                  const float* __restrict__ gate,
                                  int* __restrict__ cnt, int* __restrict__ tok,
                                  float* __restrict__ gwt) {
  int t = blockIdx.x * 8 + (threadIdx.x >> 5);
  int lane = threadIdx.x & 31;
  const float* xr = x + (size_t)t * HDIM;
  float acc[NEXP];
#pragma unroll
  for (int e = 0; e < NEXP; ++e) acc[e] = 0.f;
  for (int h = lane; h < HDIM; h += 32) {
    float xv = xr[h];
    const float* g = gate + h * NEXP;
#pragma unroll
    for (int e = 0; e < NEXP; ++e) acc[e] += xv * g[e];
  }
#pragma unroll
  for (int e = 0; e < NEXP; ++e)
    for (int off = 16; off > 0; off >>= 1) acc[e] += __shfl_xor(acc[e], off, 32);
  if (lane == 0) {
    int i0 = 0; float v0 = acc[0];
#pragma unroll
    for (int e = 1; e < NEXP; ++e) if (acc[e] > v0) { v0 = acc[e]; i0 = e; }
    int i1 = -1; float v1 = -1e30f;
#pragma unroll
    for (int e = 0; e < NEXP; ++e)
      if (e != i0 && acc[e] > v1) { v1 = acc[e]; i1 = e; }
    float w0 = 1.f / (1.f + __expf(v1 - v0));   // softmax over [v0,v1]
    float w1 = 1.f - w0;
    int p0 = atomicAdd(&cnt[i0], 1);
    tok[i0 * TMAX + p0] = t; gwt[i0 * TMAX + p0] = w0;
    int p1 = atomicAdd(&cnt[i1], 1);
    tok[i1 * TMAX + p1] = t; gwt[i1 * TMAX + p1] = w1;
  }
}

// b1[e][i] = dom[e] . W1[e][:,i] ; b3 likewise (fp32, exact)
__global__ void moe_bias_kernel(const float* __restrict__ dom,
                                const float* __restrict__ W1,
                                const float* __restrict__ W3,
                                float* __restrict__ b1, float* __restrict__ b3) {
  int g = blockIdx.x * blockDim.x + threadIdx.x;      // E*I threads
  int e = g / IDIM, i = g - e * IDIM;
  const float* d = dom + e * HDIM;
  const float* w1 = W1 + (size_t)e * HDIM * IDIM + i;
  const float* w3 = W3 + (size_t)e * HDIM * IDIM + i;
  float s1 = 0.f, s3 = 0.f;
  for (int h = 0; h < HDIM; ++h) {
    float dv = d[h];
    s1 += dv * w1[(size_t)h * IDIM];
    s3 += dv * w3[(size_t)h * IDIM];
  }
  b1[g] = s1; b3[g] = s3;
}

// pack W (E x K x N, f32 row-major) into bf16 WMMA B-fragments, frag=(nt*KT+kt)
__global__ void moe_pack_kernel(const float* __restrict__ src, uint4* __restrict__ dst,
                                int K, int N) {
  int g = blockIdx.x * blockDim.x + threadIdx.x;
  int lane = g & 31;
  int fid = g >> 5;
  int KT = K >> 5;
  int F = (N >> 4) * KT;            // frags per expert
  int e = fid / F;
  int r = fid - e * F;
  int nt = r / KT;
  int kt = r - nt * KT;
  int n  = nt * 16 + (lane & 15);
  int k0 = kt * 32 + 8 * (lane >> 4);
  const float* s = src + (size_t)e * K * N + (size_t)k0 * N + n;
  unsigned int w[8];
#pragma unroll
  for (int j = 0; j < 8; ++j) {
    float a = s[(size_t)(2 * j + (j >= 4 ? 8 : 0)) * N];
    float b = s[(size_t)(2 * j + 1 + (j >= 4 ? 8 : 0)) * N];
    w[j] = (unsigned int)f2bf(a) | ((unsigned int)f2bf(b) << 16);
  }
  uint4 lo; lo.x = w[0]; lo.y = w[1]; lo.z = w[2]; lo.w = w[3];
  uint4 hi; hi.x = w[4]; hi.y = w[5]; hi.z = w[6]; hi.w = w[7];
  uint4* d = dst + (size_t)g * 2;   // fid*64 + lane*2
  d[0] = lo; d[1] = hi;
}

// ---------------------------------------------------------------------------
// fused expert kernel: grid(64, 8) x 512 threads (16 waves)
// Wp = base of packed W1 | W3 (+PACKQ) | W2 (+2*PACKQ); bias = b1 | b3 (+E*I)
__global__ __launch_bounds__(512) void moe_expert_kernel(
    const float* __restrict__ x, const uint4* __restrict__ Wp,
    const float* __restrict__ bias, const int* __restrict__ cnt,
    const int* __restrict__ tok, const float* __restrict__ gwt,
    float* __restrict__ out) {
  const int e = blockIdx.y;
  const int n_tok = cnt[e];
  const int slot0 = blockIdx.x * MTILE;
  if (slot0 >= n_tok) return;
  const int valid = min(MTILE, n_tok - slot0);

  extern __shared__ unsigned char smem[];
  u16*   xA   = (u16*)smem;                 // [128][XSTR] bf16 activations
  u16*   hB   = xA + MTILE * XSTR;          // [128][HSTR] bf16 SwiGLU chunk
  float* gwL  = (float*)(hB + MTILE * HSTR);
  int*   tokL = (int*)(gwL + MTILE);

  const int tid = threadIdx.x;
  if (tid < MTILE) {
    int s = slot0 + min(tid, valid - 1);    // clamp tail (gate weight forced 0)
    tokL[tid] = tok[e * TMAX + s];
    gwL[tid]  = (tid < valid) ? gwt[e * TMAX + s] : 0.f;
  }
  __syncthreads();
  {  // gather 128 token rows, f32 -> bf16 into LDS (A-operand row layout)
    int r  = tid >> 2;
    int c0 = (tid & 3) * 256;
    const float* src = x + (size_t)tokL[r] * HDIM + c0;
    u16* dst = xA + r * XSTR + c0;
#pragma unroll 4
    for (int j = 0; j < 256; j += 4) {
      float4 v = *(const float4*)(src + j);
      uint2 pv;
      pv.x = (unsigned int)f2bf(v.x) | ((unsigned int)f2bf(v.y) << 16);
      pv.y = (unsigned int)f2bf(v.z) | ((unsigned int)f2bf(v.w) << 16);
      *(uint2*)(dst + j) = pv;
    }
  }

  const int wave = tid >> 5, lane = tid & 31;
  const int l15 = lane & 15, lh = lane >> 4;
  const int wn = wave & 7;    // n-role: GEMM1 n-tile / GEMM2 col block [wn*128,+128)
  const int wm = wave >> 3;   // m-half: rows [wm*64, +64)

  const size_t fsz = 64;      // uint4 per 1KB fragment
  const uint4* W1e = Wp + (size_t)e * (IDIM / 16) * KT_H * fsz;
  const uint4* W2e = Wp + 2 * PACKQ + (size_t)e * (HDIM / 16) * KT_I * fsz;
  const float* b1e = bias + e * IDIM;

  // per-m-tile LDS base pointers: all hot-loop ds offsets become short imms
  const u16 *pA[4], *pH[4];
  u16* pHs;          // h-store base
  const float* gw0;  // gate-weight base
  {
#pragma unroll
    for (int mt = 0; mt < 4; ++mt) {
      int m = wm * 64 + mt * 16 + l15;
      pA[mt] = xA + m * XSTR + 8 * lh;
      pH[mt] = hB + m * HSTR + 8 * lh;
    }
    pHs = hB + (wm * 64 + 8 * lh) * HSTR + wn * 16 + l15;
    gw0 = gwL + wm * 64 + 8 * lh;
  }

  v8f oacc[4][8];
#pragma unroll
  for (int a = 0; a < 4; ++a)
#pragma unroll
    for (int b = 0; b < 8; ++b)
#pragma unroll
      for (int q = 0; q < 8; ++q) oacc[a][b][q] = 0.f;

  __syncthreads();

  for (int c = 0; c < NCHUNKS; ++c) {
    const int ntg = c * (CCHUNK / 16) + wn;   // this wave's GEMM1 n-tile (0..255)
    v8f g1[4], g3[4];
    {
      float bv1 = b1e[ntg * 16 + l15];
      float bv3 = b1e[NEXP * IDIM + ntg * 16 + l15];
#pragma unroll
      for (int mt = 0; mt < 4; ++mt)
#pragma unroll
        for (int q = 0; q < 8; ++q) { g1[mt][q] = bv1; g3[mt][q] = bv3; }
    }
    const uint4* f1 = W1e + ((size_t)ntg * KT_H) * fsz + lane * 2;
    __builtin_prefetch((const void*)(f1 + (size_t)8 * KT_H * fsz), 0, 1);
    __builtin_prefetch((const void*)(f1 + PACKQ + (size_t)8 * KT_H * fsz), 0, 1);
#pragma unroll 2
    for (int kt = 0; kt < KT_H; ++kt) {
      v16bf B1 = load_frag_g(f1 + (size_t)kt * fsz);
      v16bf B3 = load_frag_g(f1 + PACKQ + (size_t)kt * fsz);
      v16bf A[4];
#pragma unroll
      for (int mt = 0; mt < 4; ++mt) A[mt] = load_frag_a(pA[mt], kt * 32);
#pragma unroll
      for (int mt = 0; mt < 4; ++mt) g1[mt] = wmma_b(A[mt], B1, g1[mt]);
#pragma unroll
      for (int mt = 0; mt < 4; ++mt) g3[mt] = wmma_b(A[mt], B3, g3[mt]);
    }
    __syncthreads();   // previous GEMM2 done reading hB
#pragma unroll
    for (int mt = 0; mt < 4; ++mt)
#pragma unroll
      for (int q = 0; q < 8; ++q) {
        float s1 = g1[mt][q], s3 = g3[mt][q];
        float hv = (s1 / (1.f + __expf(-s1))) * s3 * gw0[mt * 16 + q];
        pHs[(mt * 16 + q) * HSTR] = f2bf(hv);
      }
    __syncthreads();
#pragma unroll
    for (int kk = 0; kk < CCHUNK / 32; ++kk) {   // 4 k-steps over the chunk
      const int ktg = c * (CCHUNK / 32) + kk;
      const uint4* f2 = W2e + ((size_t)(wn * 8) * KT_I + ktg) * fsz + lane * 2;
      v16bf Ah[4];
#pragma unroll
      for (int mt = 0; mt < 4; ++mt) Ah[mt] = load_frag_a(pH[mt], kk * 32);
#pragma unroll
      for (int j = 0; j < 8; ++j) {
        v16bf B2 = load_frag_g(f2 + (size_t)j * KT_I * fsz);
#pragma unroll
        for (int mt = 0; mt < 4; ++mt) oacc[mt][j] = wmma_b(Ah[mt], B2, oacc[mt][j]);
      }
    }
  }
  // epilogue: exactly-2-contributions-per-element atomic add (deterministic)
#pragma unroll
  for (int mt = 0; mt < 4; ++mt)
#pragma unroll
    for (int q = 0; q < 8; ++q) {
      int m = wm * 64 + mt * 16 + q + 8 * lh;
      if (m < valid) {
        float* orow = out + (size_t)tokL[m] * HDIM + wn * 128 + l15;
#pragma unroll
        for (int j = 0; j < 8; ++j) atomicAdd(orow + j * 16, oacc[mt][j][q]);
      }
    }
}

// ---------------------------------------------------------------------------
extern "C" void kernel_launch(void* const* d_in, const int* in_sizes, int n_in,
                              void* d_out, int out_size, void* d_ws, size_t ws_size,
                              hipStream_t stream) {
  const float* x    = (const float*)d_in[0];
  const float* gate = (const float*)d_in[1];
  const float* W1   = (const float*)d_in[2];
  const float* W2   = (const float*)d_in[3];
  const float* W3   = (const float*)d_in[4];
  const float* dom  = (const float*)d_in[5];
  float* out = (float*)d_out;

  unsigned char* ws = (unsigned char*)d_ws;
  size_t o = 0;
  int*   cnt = (int*)(ws + o);   o += 256;
  int*   tok = (int*)(ws + o);   o += (size_t)NEXP * TMAX * 4;
  float* gwt = (float*)(ws + o); o += (size_t)NEXP * TMAX * 4;
  float* b1  = (float*)(ws + o); o += (size_t)NEXP * IDIM * 4;   // b3 follows b1
  float* b3  = (float*)(ws + o); o += (size_t)NEXP * IDIM * 4;
  o = (o + 255) & ~(size_t)255;
  uint4* W1p = (uint4*)(ws + o); o += (size_t)NEXP * HDIM * IDIM * 2;  // W3p, W2p follow
  uint4* W3p = W1p + PACKQ;      o += (size_t)NEXP * HDIM * IDIM * 2;
  uint4* W2p = W1p + 2 * PACKQ;  o += (size_t)NEXP * IDIM * HDIM * 2;
  // requires ws_size >= ~193 MB

  moe_zero_kernel<<<2048, 256, 0, stream>>>(out, TMAX * HDIM, cnt);
  moe_router_kernel<<<TMAX / 8, 256, 0, stream>>>(x, gate, cnt, tok, gwt);
  moe_bias_kernel<<<NEXP * IDIM / 256, 256, 0, stream>>>(dom, W1, W3, b1, b3);
  moe_pack_kernel<<<8192, 256, 0, stream>>>(W1, W1p, HDIM, IDIM);
  moe_pack_kernel<<<8192, 256, 0, stream>>>(W3, W3p, HDIM, IDIM);
  moe_pack_kernel<<<8192, 256, 0, stream>>>(W2, W2p, IDIM, HDIM);

  size_t smem = (size_t)MTILE * XSTR * 2 + (size_t)MTILE * HSTR * 2 +
                (size_t)MTILE * 4 + (size_t)MTILE * 4;   // ~300 KB of 320 KB WGP LDS
  dim3 grid(TMAX / MTILE, NEXP);
  moe_expert_kernel<<<grid, 512, smem, stream>>>(x, W1p, b1, cnt, tok, gwt, out);
}
